// Renderer_16406775071449
// MI455X (gfx1250) — matrix operand order repaired and checked by reading, not compiled, not run on hardware
//
#include <hip/hip_runtime.h>

typedef __attribute__((ext_vector_type(16))) _Float16 v16h;
typedef __attribute__((ext_vector_type(8)))  float    v8f;

#define NFREQ   512
#define FLEN    1023
#define RIRLEN  96000
#define NBATCH  8
#define NPATH   4096
#define NSURF   16
#define NFIB    128
#define NBAND   10
#define PI_F    3.14159265358979323846f

// ---------------- workspace layout (bytes) ----------------
#define WS_LR   0                                   // 16*10 f32  log(sigmoid(surface))
#define WS_SD   1024                                // 128*10 f32 sigmoid(dir)
#define WS_MH   8192                                // 512*512 f16, B-swizzled Hilbert matrix
#define WS_BAS  (8192 + 512*512*2)                  // 1024*1024 f16, B-swizzled irfft basis
#define WS_RIR  (8192 + 512*512*2 + 1024*1024*2)    // 8*96000 f32

// S(r) = sum_{n=1..511} sin(2*pi*r*n/1023); closed form, odd in r.
__device__ __forceinline__ float Sfun(int r) {
    if (r == 0) return 0.0f;
    float x = (float)r * (PI_F / 2046.0f);
    float t = tanf(x);
    if (r & 1) return 0.5f / t;     // cot
    return -0.5f * t;
}

// A-operand (16xK f16) swizzled LDS layout: element (m,k) -> half index
__device__ __forceinline__ int A_swz_idx(int m, int k) {
    int kt = k >> 5, kk = k & 31;
    int ln = m + (kk & 16);                // lane = m + 16*((kk>>4)&1)
    int j  = kk & 15;                      // j    = kk&15
    return kt * 512 + ln * 16 + j;
}

// ---------------- tiny precompute ----------------
__global__ void k_small(const float* __restrict__ sp, const float* __restrict__ dp,
                        float* __restrict__ lr, float* __restrict__ sd) {
    int t = blockIdx.x * 256 + threadIdx.x;
    if (t < NSURF * NBAND) {
        float s = 1.0f / (1.0f + expf(-sp[t]));
        lr[t] = logf(fmaxf(s, 1e-9f));
    }
    if (t < NFIB * NBAND) {
        sd[t] = 1.0f / (1.0f + expf(-dp[t]));
    }
}

// Hilbert operator Mh[q][k] (phase = log_amp @ Mh), stored B-swizzled f16.
__global__ void k_hilb(_Float16* __restrict__ MhSw) {
    int gid = blockIdx.x * 256 + threadIdx.x;     // 512*512 exact
    int q = gid >> 9, k = gid & 511;
    float v = (q == 0) ? Sfun(k) : (Sfun(k + q) + Sfun(k - q));
    v *= -2.0f / 1023.0f;
    int tk = q >> 5, kk = q & 31, tn = k >> 4, nn = k & 15;
    int ln = nn + ((kk >> 4) << 4);
    int j  = kk & 15;
    MhSw[(tk * 32 + tn) * 512 + ln * 16 + j] = (_Float16)v;
}

// irfft basis: rows 0..511 cos, rows 512..1023 (-sin), cols 0..1022 (col 1023 pad=0)
__global__ void k_bas(_Float16* __restrict__ BasSw) {
    int gid = blockIdx.x * 256 + threadIdx.x;     // 1024*1024 exact
    int r = gid >> 10, n = gid & 1023;
    float v = 0.0f;
    if (n < FLEN) {
        if (r < 512) {
            int m = (r * n) % 1023;               // exact integer angle reduction
            float sc = ((r == 0) ? 1.0f : 2.0f) / 1023.0f;
            v = sc * cosf((2.0f * PI_F / 1023.0f) * (float)m);
        } else {
            int kq = r - 512;
            if (kq > 0) {
                int m = (kq * n) % 1023;
                v = -(2.0f / 1023.0f) * sinf((2.0f * PI_F / 1023.0f) * (float)m);
            }
        }
    }
    int tk = r >> 5, kk = r & 31, tn = n >> 4, nn = n & 15;
    int ln = nn + ((kk >> 4) << 4);
    int j  = kk & 15;
    BasSw[(tk * 64 + tn) * 512 + ln * 16 + j] = (_Float16)v;
}

__global__ void k_zero(float* __restrict__ p, int n) {
    int i = blockIdx.x * 256 + threadIdx.x;
    if (i < n) p[i] = 0.0f;
}

// ---------------- main fused kernel: 64 paths per workgroup ----------------
// Four A row-tiles share every B tile load -> 1/4 the L2 B-matrix traffic,
// four independent WMMA accumulator chains per B tile. K loops capped at
// unroll_count(2) so the compiler double-buffers instead of hoisting all
// A-tiles into registers (which spilled to scratch at full unroll).
__launch_bounds__(512)
__global__ void k_main(const float* __restrict__ path_dirs,
                       const int*   __restrict__ mask,
                       const int*   __restrict__ delays,
                       const float* __restrict__ surfI,
                       const float* __restrict__ dirI,
                       const float* __restrict__ fib,
                       const float* __restrict__ log_refl,
                       const float* __restrict__ sig_dir,
                       const _Float16* __restrict__ MhSw,
                       const _Float16* __restrict__ BasSw,
                       float* __restrict__ rir) {
    __shared__ __attribute__((aligned(32))) _Float16 sh_laA[4 * 8192];    // 64KB  log_amp, A-swizzled, per row-tile
    __shared__ __attribute__((aligned(32))) _Float16 sh_fr[4 * 16384];    // 128KB [re|im], A-swizzled, per row-tile
    __shared__ float sh_lr[NSURF * NBAND];
    __shared__ float sh_sd[NFIB * NBAND];
    __shared__ float sh_fib[NFIB * 3];
    __shared__ float sh_las[64 * NBAND];
    __shared__ float sh_lad[64 * NBAND];
    __shared__ int   sh_delay[64];
    __shared__ float sh_gain[64];
    __shared__ int   sh_bidx[64];

    const int tid  = threadIdx.x;
    const int lane = tid & 31;        // gfx1250 is wave32
    const int wave = tid >> 5;        // 16 waves
    const int wg   = blockIdx.x;      // 512 WGs x 64 paths

    for (int i = tid; i < NSURF * NBAND; i += 512) sh_lr[i] = log_refl[i];
    for (int i = tid; i < NFIB * NBAND; i += 512) sh_sd[i] = sig_dir[i];
    for (int i = tid; i < NFIB * 3;     i += 512) sh_fib[i] = fib[i];
    __syncthreads();

    // ---- stage A1: per-path 10-band log-amps, delay/gain (2 waves, 64 paths) ----
    if (tid < 64) {
        int gp = wg * 64 + tid;
        float las[NBAND];
        #pragma unroll
        for (int f = 0; f < NBAND; ++f) las[f] = 0.0f;
        const int* mrow = mask + gp * NSURF;
        for (int s = 0; s < NSURF; ++s) {
            float mv = (float)mrow[s];
            #pragma unroll
            for (int f = 0; f < NBAND; ++f) las[f] += mv * sh_lr[s * NBAND + f];
        }
        float dx = path_dirs[gp * 3 + 0], dy = path_dirs[gp * 3 + 1], dz = path_dirs[gp * 3 + 2];
        float nr = sqrtf(dx * dx + dy * dy + dz * dz) + 1e-9f;
        dx /= nr; dy /= nr; dz /= nr;
        float acc[NBAND];
        #pragma unroll
        for (int f = 0; f < NBAND; ++f) acc[f] = 0.0f;
        float wsum = 0.0f;
        for (int n = 0; n < NFIB; ++n) {
            float dot = dx * sh_fib[n * 3] + dy * sh_fib[n * 3 + 1] + dz * sh_fib[n * 3 + 2];
            float e = expf(8.0f * dot);
            wsum += e;
            #pragma unroll
            for (int f = 0; f < NBAND; ++f) acc[f] += e * sh_sd[n * NBAND + f];
        }
        #pragma unroll
        for (int f = 0; f < NBAND; ++f) {
            sh_las[tid * NBAND + f] = las[f];
            sh_lad[tid * NBAND + f] = logf(fmaxf(acc[f] / wsum, 1e-9f));
        }
        int d0 = delays[gp];
        sh_delay[tid] = d0;
        sh_gain[tid]  = 1.0f / fmaxf((float)d0 / 48.0f, 1.0f);
        sh_bidx[tid]  = gp >> 12;     // P = 4096
    }
    __syncthreads();

    // ---- stage A2: interpolate to 512 bins; interp values loaded once per bin ----
    for (int q = tid; q < NFREQ; q += 512) {
        float si[NBAND], di[NBAND];
        #pragma unroll
        for (int f = 0; f < NBAND; ++f) {
            si[f] = surfI[f * NFREQ + q];
            di[f] = dirI[f * NFREQ + q];
        }
        for (int m = 0; m < 64; ++m) {
            float s = 0.0f;
            #pragma unroll
            for (int f = 0; f < NBAND; ++f)
                s += sh_las[m * NBAND + f] * si[f] + sh_lad[m * NBAND + f] * di[f];
            sh_laA[(m >> 4) * 8192 + A_swz_idx(m & 15, q)] = (_Float16)s;
        }
    }
    __syncthreads();

    // ---- stage B: phase = log_amp @ Mh (WMMA, B shared by 4 row tiles) ----
    // ---- stage C: fr = exp(la) * e^{-i*phase}, written A-swizzled to sh_fr ----
    for (int t = 0; t < 2; ++t) {
        int tn = wave * 2 + t;                 // 32 column tiles of 16
        v8f c[4] = {{}, {}, {}, {}};
        #pragma clang loop unroll_count(2)
        for (int kt = 0; kt < 16; ++kt) {
            v16h b = *(const v16h*)&MhSw[(kt * 32 + tn) * 512 + lane * 16];
            #pragma unroll
            for (int rt = 0; rt < 4; ++rt) {
                v16h a = *(const v16h*)&sh_laA[rt * 8192 + kt * 512 + lane * 16];
                c[rt] = __builtin_amdgcn_wmma_f32_16x16x32_f16(false, a, false, b, (short)0, c[rt], false, false);
            }
        }
        int n  = tn * 16 + (lane & 15);
        int mb = (lane >> 4) * 8;
        #pragma unroll
        for (int rt = 0; rt < 4; ++rt) {
            #pragma unroll
            for (int i = 0; i < 8; ++i) {
                int mm = mb + i;
                float la = (float)sh_laA[rt * 8192 + A_swz_idx(mm, n)];
                float A  = expf(la);
                float sn, cs;
                sincosf(c[rt][i], &sn, &cs);
                sh_fr[rt * 16384 + A_swz_idx(mm, n)]       = (_Float16)(A * cs);
                sh_fr[rt * 16384 + A_swz_idx(mm, 512 + n)] = (_Float16)(-A * sn);
            }
        }
    }
    __syncthreads();

    // ---- stage D: filt = [re|im] @ Bas (WMMA, shared B); stage E: scatter ----
    for (int t = 0; t < 4; ++t) {
        int tn = wave * 4 + t;                 // 64 column tiles of 16
        v8f c[4] = {{}, {}, {}, {}};
        #pragma clang loop unroll_count(2)
        for (int kt = 0; kt < 32; ++kt) {
            v16h b = *(const v16h*)&BasSw[(kt * 64 + tn) * 512 + lane * 16];
            #pragma unroll
            for (int rt = 0; rt < 4; ++rt) {
                v16h a = *(const v16h*)&sh_fr[rt * 16384 + kt * 512 + lane * 16];
                c[rt] = __builtin_amdgcn_wmma_f32_16x16x32_f16(false, a, false, b, (short)0, c[rt], false, false);
            }
        }
        int n = tn * 16 + (lane & 15);
        if (n < FLEN) {
            int mb = (lane >> 4) * 8;
            #pragma unroll
            for (int rt = 0; rt < 4; ++rt) {
                #pragma unroll
                for (int i = 0; i < 8; ++i) {
                    int m = rt * 16 + mb + i;      // path within WG
                    atomicAdd(&rir[sh_bidx[m] * RIRLEN + sh_delay[m] + n], c[rt][i] * sh_gain[m]);
                }
            }
        }
    }
}

// ---------------- 'same' convolution with source kernel ----------------
__launch_bounds__(256)
__global__ void k_conv(const float* __restrict__ rir, const float* __restrict__ sk,
                       float* __restrict__ out) {
    __shared__ float shk[FLEN];
    __shared__ float shr[2046];
    int bidx = blockIdx.x / 94;
    int t0   = (blockIdx.x % 94) * 1024;
    for (int i = threadIdx.x; i < FLEN; i += 256) shk[i] = sk[i];
    for (int i = threadIdx.x; i < 2046; i += 256) {
        int g = t0 - 511 + i;
        shr[i] = (g >= 0 && g < RIRLEN) ? rir[bidx * RIRLEN + g] : 0.0f;
    }
    __syncthreads();
    float a0 = 0.0f, a1 = 0.0f, a2 = 0.0f, a3 = 0.0f;
    int base = threadIdx.x;
    for (int j = 0; j < FLEN; ++j) {
        float kv = shk[j];
        a0 += shr[base + j] * kv;
        a1 += shr[base + 256 + j] * kv;
        a2 += shr[base + 512 + j] * kv;
        a3 += shr[base + 768 + j] * kv;
    }
    int t = t0 + threadIdx.x;
    if (t < RIRLEN)        out[bidx * RIRLEN + t]        = a0;
    if (t + 256 < RIRLEN)  out[bidx * RIRLEN + t + 256]  = a1;
    if (t + 512 < RIRLEN)  out[bidx * RIRLEN + t + 512]  = a2;
    if (t + 768 < RIRLEN)  out[bidx * RIRLEN + t + 768]  = a3;
}

extern "C" void kernel_launch(void* const* d_in, const int* in_sizes, int n_in,
                              void* d_out, int out_size, void* d_ws, size_t ws_size,
                              hipStream_t stream) {
    const float* surface_params = (const float*)d_in[0];   // [16,10]
    const float* dir_params     = (const float*)d_in[1];   // [128,10]
    const float* path_dirs      = (const float*)d_in[2];   // [8,4096,3]
    const float* source_kernel  = (const float*)d_in[3];   // [1023]
    const float* surf_interp    = (const float*)d_in[4];   // [10,512]
    const float* dir_interp     = (const float*)d_in[5];   // [10,512]
    const float* fib_points     = (const float*)d_in[6];   // [128,3]
    const int*   mask           = (const int*)d_in[7];     // [8,4096,16]
    const int*   delays         = (const int*)d_in[8];     // [8,4096]
    float* out = (float*)d_out;

    char* ws = (char*)d_ws;
    float*    lr  = (float*)(ws + WS_LR);
    float*    sd  = (float*)(ws + WS_SD);
    _Float16* Mh  = (_Float16*)(ws + WS_MH);
    _Float16* Bas = (_Float16*)(ws + WS_BAS);
    float*    rir = (float*)(ws + WS_RIR);

    k_small<<<5, 256, 0, stream>>>(surface_params, dir_params, lr, sd);
    k_hilb<<<1024, 256, 0, stream>>>(Mh);       // 512*512
    k_bas<<<4096, 256, 0, stream>>>(Bas);       // 1024*1024
    k_zero<<<(NBATCH * RIRLEN + 255) / 256, 256, 0, stream>>>(rir, NBATCH * RIRLEN);
    k_main<<<512, 512, 0, stream>>>(path_dirs, mask, delays, surf_interp, dir_interp,
                                    fib_points, lr, sd, Mh, Bas, rir);
    k_conv<<<NBATCH * 94, 256, 0, stream>>>(rir, source_kernel, out);
}
